// MultiHeadSelfAttention_65996467470473
// MI455X (gfx1250) — matrix operand order repaired and checked by reading, not compiled, not run on hardware
//
#include <hip/hip_runtime.h>
#include <hip/hip_bf16.h>
#include <float.h>

// ---------------- problem constants ----------------
#define BATCH   2
#define SEQ     2048
#define DMODEL  1024
#define HIDSZ   1024
#define NHEAD   16
#define DHEAD   64
#define NROWS   (BATCH * SEQ)   // 4096

typedef __bf16 bf16_t;
typedef __attribute__((ext_vector_type(8)))  __bf16 v8bf;
typedef __attribute__((ext_vector_type(16))) __bf16 v16bf;
typedef __attribute__((ext_vector_type(8)))  float  v8f;

// ---------------- WMMA helpers (contiguous k order: lane half h holds k = 16h..16h+15, wave32) ----------------

// A-matrix 16x32 bf16: lanes 0..15: M=lane, K=[0..16); lanes 16..31: M=lane-16, K=[16..32)
__device__ __forceinline__ v16bf load_a_bf16(const bf16_t* __restrict__ base, int ld,
                                             int row0, int k0) {
  const int lane = threadIdx.x & 31;
  const int m    = lane & 15;
  const int koff = (lane < 16) ? 0 : 16;
  const bf16_t* p = base + (size_t)(row0 + m) * ld + (k0 + koff);
  v8bf lo = *(const v8bf*)(p);
  v8bf hi = *(const v8bf*)(p + 8);
  v16bf a;
#pragma unroll
  for (int i = 0; i < 8; ++i) { a[i] = lo[i]; a[8 + i] = hi[i]; }
  return a;
}

// B-matrix 32x16 bf16 with B[k,n] = Row(n)[k]: lanes 0..15: N=lane, K=[0..16); lanes 16..31: K=[16..32)
__device__ __forceinline__ v16bf load_b_bf16(const bf16_t* __restrict__ base, int ld,
                                             int col0, int k0) {
  const int lane = threadIdx.x & 31;
  const int n    = lane & 15;
  const int koff = (lane < 16) ? 0 : 16;
  const bf16_t* p = base + (size_t)(col0 + n) * ld + (k0 + koff);
  return *(const v16bf*)p;
}

__device__ __forceinline__ v8f wmma_bf16(v16bf a, v16bf b, v8f c) {
  return __builtin_amdgcn_wmma_f32_16x16x32_bf16(false, a, false, b, (short)0, c,
                                                 false, false);
}

// ---------------- kernels ----------------

__global__ void cvt_f32_bf16(const float* __restrict__ in, bf16_t* __restrict__ out, int n) {
  int i = blockIdx.x * blockDim.x + threadIdx.x;
  if (i < n) out[i] = (bf16_t)in[i];
}

// Q/K/V projections. 4x4 register blocking: one wave computes a 64x64 output tile.
// __launch_bounds__(32, 1): single-wave workgroup, min-waves-per-EU = 1, so the
// compiler may use the full VGPR budget (16 v8f accumulators + 4 A-tiles resident,
// no scratch spills). V is stored transposed: vt[col][row].
__global__ void __launch_bounds__(32, 1) qkv_gemm(
    const bf16_t* __restrict__ xb,
    const bf16_t* __restrict__ wq,
    const bf16_t* __restrict__ wk,
    const bf16_t* __restrict__ wv,
    bf16_t* __restrict__ q,
    bf16_t* __restrict__ k,
    bf16_t* __restrict__ vt) {
  const int j0    = blockIdx.x * 64;
  const int i0    = blockIdx.y * 64;
  const int which = blockIdx.z;
  const bf16_t* w = (which == 0) ? wq : (which == 1) ? wk : wv;

  v8f c[4][4] = {};
  for (int k0 = 0; k0 < DMODEL; k0 += 32) {
    __builtin_prefetch(w + (size_t)j0 * DMODEL + k0 + 128, 0, 1);
    v16bf a[4];
#pragma unroll
    for (int ii = 0; ii < 4; ++ii) a[ii] = load_a_bf16(xb, DMODEL, i0 + ii * 16, k0);
#pragma unroll
    for (int jj = 0; jj < 4; ++jj) {
      v16bf bm = load_b_bf16(w, DMODEL, j0 + jj * 16, k0);
#pragma unroll
      for (int ii = 0; ii < 4; ++ii) c[ii][jj] = wmma_bf16(a[ii], bm, c[ii][jj]);
    }
  }
  const int lane = threadIdx.x & 31;
  const int n    = lane & 15;
  const int moff = (lane < 16) ? 0 : 8;
  if (which == 2) {
#pragma unroll
    for (int ii = 0; ii < 4; ++ii)
#pragma unroll
      for (int jj = 0; jj < 4; ++jj)
#pragma unroll
        for (int r = 0; r < 8; ++r)
          vt[(size_t)(j0 + jj * 16 + n) * NROWS + (i0 + ii * 16 + moff + r)] =
              (bf16_t)c[ii][jj][r];
  } else {
    bf16_t* dst = (which == 0) ? q : k;
#pragma unroll
    for (int ii = 0; ii < 4; ++ii)
#pragma unroll
      for (int jj = 0; jj < 4; ++jj)
#pragma unroll
        for (int r = 0; r < 8; ++r)
          dst[(size_t)(i0 + ii * 16 + moff + r) * HIDSZ + (j0 + jj * 16 + n)] =
              (bf16_t)c[ii][jj][r];
  }
}

// Fused attention: scores (Q.K^T) + mask + softmax + attn write + attn@V.
// One workgroup = 16 waves (512 thr) per (bh, 32-row block: 2 i-tiles).
// Wave w owns columns [w*128, w*128+128); K/V B-tiles are loaded once and
// shared by both i-tiles' WMMAs. Scores live in 2x8 v8f accumulators (128 VGPRs).
// Plain __launch_bounds__(512): 4 waves/SIMD -> 256-VGPR cap, which the ~224-VGPR
// working set fits under while keeping the 16-wave block launchable.
__global__ void __launch_bounds__(512) fused_attn(
    const bf16_t* __restrict__ q,
    const bf16_t* __restrict__ kk,
    const bf16_t* __restrict__ vt,
    const unsigned char* __restrict__ mask,
    float* __restrict__ attn,
    bf16_t* __restrict__ ob) {
  const int i0 = blockIdx.x * 32;
  const int bh = blockIdx.y;
  const int b  = bh >> 4;
  const int h  = bh & 15;
  const int tid  = threadIdx.x;
  const int w    = tid >> 5;       // 0..15
  const int lane = tid & 31;
  const int n    = lane & 15;
  const int moff = (lane < 16) ? 0 : 8;

  __shared__ float  wstat[16][32];        // per-wave row partials (max, then sum)
  __shared__ float  rowmax_s[32];
  __shared__ float  rowinv_s[32];
  __shared__ float  obuf[32 * DHEAD];     // 8KB O accumulator
  __shared__ bf16_t pbuf[16][2][16 * 32]; // per-wave prob-tile transpose staging (32KB)

  for (int t = tid; t < 32 * DHEAD; t += 512) obuf[t] = 0.f;

  const bf16_t* qbase = q  + (size_t)b * SEQ * HIDSZ + h * DHEAD;
  const bf16_t* kbase = kk + (size_t)b * SEQ * HIDSZ + h * DHEAD;

  // persistent Q A-tiles for both i-tiles (K = 0..32, 32..64)
  v16bf qa[2][2];
#pragma unroll
  for (int it = 0; it < 2; ++it) {
    qa[it][0] = load_a_bf16(qbase, HIDSZ, i0 + it * 16, 0);
    qa[it][1] = load_a_bf16(qbase, HIDSZ, i0 + it * 16, 32);
  }

  const int jbase = w * 128;
  v8f c[2][8];

  // ---- phase 1: scores + mask (K tiles loaded once, used by both i-tiles) ----
#pragma unroll
  for (int jt = 0; jt < 8; ++jt) {
    const int j0 = jbase + jt * 16;
    v16bf b0 = load_b_bf16(kbase, HIDSZ, j0, 0);
    v16bf b1 = load_b_bf16(kbase, HIDSZ, j0, 32);
    const bool mv = mask[(size_t)b * SEQ + j0 + n] != 0;
#pragma unroll
    for (int it = 0; it < 2; ++it) {
      v8f acc = {};
      acc = wmma_bf16(qa[it][0], b0, acc);
      acc = wmma_bf16(qa[it][1], b1, acc);
#pragma unroll
      for (int r = 0; r < 8; ++r) acc[r] = mv ? acc[r] : -FLT_MAX;
      c[it][jt] = acc;
    }
  }

  // ---- phase 2: row max ----
#pragma unroll
  for (int it = 0; it < 2; ++it) {
    float rmax[8];
#pragma unroll
    for (int r = 0; r < 8; ++r) rmax[r] = -FLT_MAX;
#pragma unroll
    for (int jt = 0; jt < 8; ++jt)
#pragma unroll
      for (int r = 0; r < 8; ++r) rmax[r] = fmaxf(rmax[r], c[it][jt][r]);
#pragma unroll
    for (int msk = 8; msk >= 1; msk >>= 1)
#pragma unroll
      for (int r = 0; r < 8; ++r) rmax[r] = fmaxf(rmax[r], __shfl_xor(rmax[r], msk, 32));
    if (n == 0) {  // lane 0: rows +0..7, lane 16: rows +8..15
#pragma unroll
      for (int r = 0; r < 8; ++r) wstat[w][it * 16 + moff + r] = rmax[r];
    }
  }
  __syncthreads();
  if (tid < 32) {
    float m = wstat[0][tid];
#pragma unroll
    for (int ww = 1; ww < 16; ++ww) m = fmaxf(m, wstat[ww][tid]);
    rowmax_s[tid] = m;
  }
  __syncthreads();

  // ---- phase 3: exp + row sum ----
#pragma unroll
  for (int it = 0; it < 2; ++it) {
    float rowm[8], rsum[8];
#pragma unroll
    for (int r = 0; r < 8; ++r) { rowm[r] = rowmax_s[it * 16 + moff + r]; rsum[r] = 0.f; }
#pragma unroll
    for (int jt = 0; jt < 8; ++jt)
#pragma unroll
      for (int r = 0; r < 8; ++r) {
        float e = __expf(c[it][jt][r] - rowm[r]);
        c[it][jt][r] = e;
        rsum[r] += e;
      }
#pragma unroll
    for (int msk = 8; msk >= 1; msk >>= 1)
#pragma unroll
      for (int r = 0; r < 8; ++r) rsum[r] += __shfl_xor(rsum[r], msk, 32);
    if (n == 0) {
#pragma unroll
      for (int r = 0; r < 8; ++r) wstat[w][it * 16 + moff + r] = rsum[r];
    }
  }
  __syncthreads();
  if (tid < 32) {
    float s = wstat[0][tid];
#pragma unroll
    for (int ww = 1; ww < 16; ++ww) s += wstat[ww][tid];
    rowinv_s[tid] = 1.0f / s;
  }
  __syncthreads();
  float rinv[2][8];
#pragma unroll
  for (int it = 0; it < 2; ++it)
#pragma unroll
    for (int r = 0; r < 8; ++r) rinv[it][r] = rowinv_s[it * 16 + moff + r];

  // ---- phase 4: normalize, write attn, accumulate O = P @ V (V tiles shared) ----
  float* abase = attn + (size_t)bh * SEQ * SEQ;
  const int koffA = (lane < 16) ? 0 : 16;
  const int koffB = (lane < 16) ? 0 : 16;
  const int m     = lane & 15;
  v8f co[2][4] = {};
#pragma unroll
  for (int chunk = 0; chunk < 4; ++chunk) {
    v16bf a[2];
#pragma unroll
    for (int it = 0; it < 2; ++it) {
#pragma unroll
      for (int half = 0; half < 2; ++half) {
        const int jt = chunk * 2 + half;
        const int j0 = jbase + jt * 16;
#pragma unroll
        for (int r = 0; r < 8; ++r) {
          float p = c[it][jt][r] * rinv[it][r];
          abase[(size_t)(i0 + it * 16 + moff + r) * SEQ + j0 + n] = p;  // attn output
          pbuf[w][it][(moff + r) * 32 + half * 16 + n] = (bf16_t)p;     // transpose stage
        }
      }
      // rebuild 16x32 prob tile in A layout (same-wave DS ops are in-order)
      v8bf lo0 = *(const v8bf*)&pbuf[w][it][m * 32 + koffA];
      v8bf hi0 = *(const v8bf*)&pbuf[w][it][m * 32 + 8 + koffA];
#pragma unroll
      for (int e = 0; e < 8; ++e) { a[it][e] = lo0[e]; a[it][8 + e] = hi0[e]; }
    }
    const int kj = b * SEQ + jbase + chunk * 32;
#pragma unroll
    for (int nt = 0; nt < 4; ++nt) {
      const bf16_t* pb = vt + (size_t)(h * DHEAD + nt * 16 + n) * NROWS + kj + koffB;
      v16bf bv = *(const v16bf*)pb;
#pragma unroll
      for (int it = 0; it < 2; ++it) co[it][nt] = wmma_bf16(a[it], bv, co[it][nt]);
    }
  }

  // ---- phase 5: cross-wave O reduction (LDS float atomics) + store ----
#pragma unroll
  for (int it = 0; it < 2; ++it)
#pragma unroll
    for (int nt = 0; nt < 4; ++nt)
#pragma unroll
      for (int r = 0; r < 8; ++r)
        atomicAdd(&obuf[(it * 16 + moff + r) * DHEAD + nt * 16 + n], co[it][nt][r]);
  __syncthreads();
  const float scale = 0.03125f;  // HID^-0.5 = 1/32
  for (int t = tid; t < 32 * DHEAD; t += 512) {
    const int row = t >> 6;
    const int col = t & 63;
    ob[(size_t)(b * SEQ + i0 + row) * HIDSZ + h * DHEAD + col] = (bf16_t)(obuf[t] * scale);
  }
}

// out = O Wo^T, 4x4 register blocking (64x64 per wave), full VGPR budget
__global__ void __launch_bounds__(32, 1) out_gemm(
    const bf16_t* __restrict__ ob,
    const bf16_t* __restrict__ wo,
    float* __restrict__ out) {
  const int j0 = blockIdx.x * 64;
  const int i0 = blockIdx.y * 64;
  v8f c[4][4] = {};
  for (int k0 = 0; k0 < HIDSZ; k0 += 32) {
    __builtin_prefetch(wo + (size_t)j0 * HIDSZ + k0 + 128, 0, 1);
    v16bf a[4];
#pragma unroll
    for (int ii = 0; ii < 4; ++ii) a[ii] = load_a_bf16(ob, HIDSZ, i0 + ii * 16, k0);
#pragma unroll
    for (int jj = 0; jj < 4; ++jj) {
      v16bf bm = load_b_bf16(wo, HIDSZ, j0 + jj * 16, k0);
#pragma unroll
      for (int ii = 0; ii < 4; ++ii) c[ii][jj] = wmma_bf16(a[ii], bm, c[ii][jj]);
    }
  }
  const int lane = threadIdx.x & 31;
  const int n    = lane & 15;
  const int moff = (lane < 16) ? 0 : 8;
#pragma unroll
  for (int ii = 0; ii < 4; ++ii)
#pragma unroll
    for (int jj = 0; jj < 4; ++jj)
#pragma unroll
      for (int r = 0; r < 8; ++r)
        out[(size_t)(i0 + ii * 16 + moff + r) * DMODEL + (j0 + jj * 16 + n)] = c[ii][jj][r];
}

// ---------------- host launcher ----------------
extern "C" void kernel_launch(void* const* d_in, const int* in_sizes, int n_in,
                              void* d_out, int out_size, void* d_ws, size_t ws_size,
                              hipStream_t stream) {
  (void)in_sizes; (void)n_in; (void)out_size; (void)ws_size;
  const float*         x    = (const float*)d_in[0];
  const unsigned char* mask = (const unsigned char*)d_in[1];
  const float*         Wq   = (const float*)d_in[2];
  const float*         Wk   = (const float*)d_in[3];
  const float*         Wv   = (const float*)d_in[4];
  const float*         Wo   = (const float*)d_in[5];

  char* ws = (char*)d_ws;
  bf16_t* xb  = (bf16_t*)ws; ws += (size_t)NROWS * DMODEL * sizeof(bf16_t);
  bf16_t* wqb = (bf16_t*)ws; ws += (size_t)HIDSZ * DMODEL * sizeof(bf16_t);
  bf16_t* wkb = (bf16_t*)ws; ws += (size_t)HIDSZ * DMODEL * sizeof(bf16_t);
  bf16_t* wvb = (bf16_t*)ws; ws += (size_t)HIDSZ * DMODEL * sizeof(bf16_t);
  bf16_t* wob = (bf16_t*)ws; ws += (size_t)DMODEL * HIDSZ * sizeof(bf16_t);
  bf16_t* qb  = (bf16_t*)ws; ws += (size_t)NROWS * HIDSZ * sizeof(bf16_t);
  bf16_t* kb  = (bf16_t*)ws; ws += (size_t)NROWS * HIDSZ * sizeof(bf16_t);
  bf16_t* vtb = (bf16_t*)ws; ws += (size_t)HIDSZ * NROWS * sizeof(bf16_t);
  bf16_t* obuf= (bf16_t*)ws; ws += (size_t)NROWS * HIDSZ * sizeof(bf16_t);

  float* outp = (float*)d_out;                  // (B,S,D) = 4,194,304 f32
  float* attn = outp + (size_t)NROWS * DMODEL;  // (B,H,S,S) follows

  {
    int n = NROWS * DMODEL;
    cvt_f32_bf16<<<dim3((n + 255) / 256), dim3(256), 0, stream>>>(x, xb, n);
    n = HIDSZ * DMODEL;
    cvt_f32_bf16<<<dim3((n + 255) / 256), dim3(256), 0, stream>>>(Wq, wqb, n);
    cvt_f32_bf16<<<dim3((n + 255) / 256), dim3(256), 0, stream>>>(Wk, wkb, n);
    cvt_f32_bf16<<<dim3((n + 255) / 256), dim3(256), 0, stream>>>(Wv, wvb, n);
    cvt_f32_bf16<<<dim3((n + 255) / 256), dim3(256), 0, stream>>>(Wo, wob, n);
  }
  qkv_gemm<<<dim3(HIDSZ / 64, NROWS / 64, 3), dim3(32), 0, stream>>>(
      xb, wqb, wkb, wvb, qb, kb, vtb);
  fused_attn<<<dim3(SEQ / 32, BATCH * NHEAD), dim3(512), 0, stream>>>(
      qb, kb, vtb, mask, attn, obuf);
  out_gemm<<<dim3(DMODEL / 64, NROWS / 64, 1), dim3(32), 0, stream>>>(obuf, wob, outp);
}
